// DecisionTransformer_36670430773432
// MI455X (gfx1250) — matrix operand order repaired and checked by reading, not compiled, not run on hardware
//
#include <hip/hip_runtime.h>
#include <math.h>

// ---------------- problem constants ----------------
#define B_ 4
#define T_ 512
#define S_ 1536          // 3*T
#define H_ 1024
#define NH_ 16
#define D_ 64
#define L_ 4
#define FF_ 4096
#define MROWS_ (B_ * S_) // 6144
#define LN_EPS_ 1e-5f

typedef __attribute__((ext_vector_type(16))) __bf16 v16bf;
typedef __attribute__((ext_vector_type(8)))  __bf16 v8bf;
typedef __attribute__((ext_vector_type(8)))  float  v8f;
typedef __attribute__((ext_vector_type(4)))  unsigned int u32x4;
typedef __attribute__((ext_vector_type(8)))  int i32x8;
typedef __attribute__((ext_vector_type(4)))  int i32x4;

// LDS row pitch for staged tiles: 32 bf16 (64B) + 16B TDM pad -> 80B (40 elems)
#define LDSPITCH_ 40

// ---------------------------------------------------------------------------
// Tensor Data Mover: DMA a 2-D tile (rows x 32 bf16, row stride K elements)
// from global memory into LDS at lds_addr (bytes), padding 16B per 64B row.
// Tracked with TENSORcnt.
// ---------------------------------------------------------------------------
#if defined(__has_builtin)
#if __has_builtin(__builtin_amdgcn_tensor_load_to_lds)
#define HAVE_TDM_ 1
#endif
#endif

__device__ __forceinline__ void tdm_load_tile(unsigned lds_addr,
                                              const __bf16* gsrc,
                                              int rows, int strideK) {
#ifdef HAVE_TDM_
    const unsigned long long ga = (unsigned long long)(uintptr_t)gsrc;
    u32x4 g0;
    g0[0] = 1u;                                        // count=1, user mode
    g0[1] = lds_addr;                                  // LDS byte address
    g0[2] = (unsigned)(ga & 0xffffffffu);              // global_addr[31:0]
    g0[3] = (unsigned)((ga >> 32) & 0x01ffffffu)       // global_addr[56:32]
            | 0x80000000u;                             // type = 2 ("image")
    i32x8 g1;
    g1[0] = (int)((1u << 16)                           // data_size = 2 bytes
                  | (1u << 20)                         // pad_enable
                  | (3u << 22)                         // pad_interval: 16 DWORDs
                  | (3u << 25));                       // pad_amount: 4 DWORDs
    g1[1] = (int)(32u << 16);                          // tensor_dim0 = 32 (lo16)
    g1[2] = (int)((unsigned)rows << 16);               // dim0 hi=0 | tensor_dim1 lo16
    g1[3] = (int)(32u << 16);                          // dim1 hi=0 | tile_dim0 = 32
    g1[4] = rows;                                      // tile_dim1 = rows, tile_dim2=0
    g1[5] = strideK;                                   // tensor_dim0_stride lo32
    g1[6] = 0;
    g1[7] = 0;
    const i32x4 z4 = {0, 0, 0, 0};
#if __has_include(<hip/amd_detail/amd_gfx1250_TDM.h>)
    const i32x8 z8 = {0, 0, 0, 0, 0, 0, 0, 0};
    __builtin_amdgcn_tensor_load_to_lds(g0, g1, z4, z4, z8, 0);   // clang-23 form
#else
    __builtin_amdgcn_tensor_load_to_lds(g0, g1, z4, z4, 0);       // ROCm 7.2 form
#endif
#else
    (void)gsrc; (void)rows; (void)strideK; (void)lds_addr;
#endif
}

template <int NWAIT>
__device__ __forceinline__ void tdm_wait() {
#ifdef HAVE_TDM_
    __builtin_amdgcn_s_wait_tensorcnt(NWAIT);   // constant integer required
#endif
}

// ---------------------------------------------------------------------------
// block-wide sum over 256 threads (8 waves, wave32)
// ---------------------------------------------------------------------------
__device__ __forceinline__ float block_sum256(float v) {
    __shared__ float sred[8];
    #pragma unroll
    for (int off = 16; off > 0; off >>= 1) v += __shfl_down(v, off, 32);
    if ((threadIdx.x & 31) == 0) sred[threadIdx.x >> 5] = v;
    __syncthreads();
    float tot = sred[0] + sred[1] + sred[2] + sred[3] +
                sred[4] + sred[5] + sred[6] + sred[7];
    __syncthreads();
    return tot;
}

// ---------------------------------------------------------------------------
// Embedding + interleave + LayerNorm.  One block (256 thr) per token row.
// ---------------------------------------------------------------------------
__global__ void embed_ln_kernel(const float* __restrict__ states,
                                const float* __restrict__ rtg,
                                const int*   __restrict__ timesteps,
                                const int*   __restrict__ actions,
                                const float* __restrict__ tt,
                                const float* __restrict__ at,
                                const float* __restrict__ Wrtg,
                                const float* __restrict__ brtg,
                                const float* __restrict__ Wst,
                                const float* __restrict__ bst,
                                float* __restrict__ hF,
                                __bf16* __restrict__ hB) {
    const int row  = blockIdx.x;
    const int b    = row / S_;
    const int sidx = row - b * S_;
    const int t    = sidx / 3;
    const int type = sidx - 3 * t;
    const int tid  = threadIdx.x;

    const int ts = timesteps[b * T_ + t];
    const float* te = tt + (long)ts * H_;

    float x[4];
    #pragma unroll
    for (int u = 0; u < 4; ++u) {
        const int j = tid + u * 256;
        float v;
        if (type == 0) {
            v = rtg[b * T_ + t] * Wrtg[j] + brtg[j];
        } else if (type == 1) {
            v = bst[j];
            const float* st = states + (long)(b * T_ + t) * 17;
            #pragma unroll
            for (int i = 0; i < 17; ++i) v += st[i] * Wst[i * H_ + j];
        } else {
            v = at[(long)actions[b * T_ + t] * H_ + j];
        }
        x[u] = v + te[j];
    }
    float s = x[0] + x[1] + x[2] + x[3];
    const float mean = block_sum256(s) * (1.0f / H_);
    float vs = 0.f;
    #pragma unroll
    for (int u = 0; u < 4; ++u) { float d = x[u] - mean; vs += d * d; }
    const float var = block_sum256(vs) * (1.0f / H_);
    const float rs  = rsqrtf(var + LN_EPS_);
    #pragma unroll
    for (int u = 0; u < 4; ++u) {
        const int j = tid + u * 256;
        const float o = (x[u] - mean) * rs;
        hF[(long)row * H_ + j] = o;
        hB[(long)row * H_ + j] = (__bf16)o;
    }
}

// ---------------------------------------------------------------------------
// fp32 (K x N) weight -> bf16 transposed (N x K)
// ---------------------------------------------------------------------------
__global__ void conv_transpose_kernel(const float* __restrict__ in,
                                      __bf16* __restrict__ out, int K, int N) {
    const long idx = (long)blockIdx.x * 256 + threadIdx.x;
    if (idx >= (long)K * N) return;
    const int k = (int)(idx / N);
    const int n = (int)(idx - (long)k * N);
    out[(long)n * K + k] = (__bf16)in[idx];
}

// ---------------------------------------------------------------------------
// v (MROWS x H bf16) -> vT[b][head][d][S] bf16
// ---------------------------------------------------------------------------
__global__ void transpose_v_kernel(const __bf16* __restrict__ v,
                                   __bf16* __restrict__ vT) {
    const long idx = (long)blockIdx.x * 256 + threadIdx.x;
    if (idx >= (long)MROWS_ * H_) return;
    const int row = (int)(idx / H_);
    const int c   = (int)(idx - (long)row * H_);
    const int b   = row / S_;
    const int s   = row - b * S_;
    const int hd  = c / D_;
    const int d   = c - hd * D_;
    vT[(((long)(b * NH_ + hd)) * D_ + d) * S_ + s] = v[idx];
}

// ---------------------------------------------------------------------------
// GEMM: C[M,N] = A[M,K](bf16) x W (via WT[N,K] bf16) + bias[n]
// MODE 0: fp32 out, 1: bf16 out, 2: bf16 out + exact GELU.
// Block = 256 threads (8 waves, 4x2); block tile 256(M) x 128(N); each wave
// computes 64x64 = 16 v_wmma_f32_16x16x32_bf16 per K-step of 32.
// A/B tiles are staged into LDS by the Tensor Data Mover (double buffered,
// TDM row padding 64B->80B to dodge ds_load_b128 bank conflicts).
// ---------------------------------------------------------------------------
template <int MODE>
__global__ void gemm_wmma_kernel(const __bf16* __restrict__ A,
                                 const __bf16* __restrict__ WT,
                                 const float* __restrict__ bias,
                                 void* __restrict__ Cout,
                                 int M, int N, int K) {
    __shared__ __bf16 sA[2][256 * LDSPITCH_];   // 2 x 20KB
    __shared__ __bf16 sB[2][128 * LDSPITCH_];   // 2 x 10KB

    const int tid  = threadIdx.x;
    const int wave = tid >> 5;
    const int lane = tid & 31;
    const int wm = wave >> 1, wn = wave & 1;            // 4 x 2 wave grid
    const long mblock = (long)blockIdx.y * 256;
    const long nblock = (long)blockIdx.x * 128;
    const int hh  = lane >> 4;
    const int l15 = lane & 15;

    v8f acc[4][4];
    #pragma unroll
    for (int i = 0; i < 4; ++i)
        #pragma unroll
        for (int j = 0; j < 4; ++j)
            acc[i][j] = (v8f){0.f, 0.f, 0.f, 0.f, 0.f, 0.f, 0.f, 0.f};

#ifdef HAVE_TDM_
    // prologue: DMA first K-slab into buffer 0 (wave 0 only; TDM ignores EXEC)
    if (tid < 32) {
        tdm_load_tile((unsigned)(uintptr_t)&sA[0][0], A + mblock * K, 256, K);
        tdm_load_tile((unsigned)(uintptr_t)&sB[0][0], WT + nblock * K, 128, K);
    }
#endif

    int ibuf = 0;
    for (int k0 = 0; k0 < K; k0 += 32) {
#ifdef HAVE_TDM_
        if (tid < 32) {
            if (k0 + 32 < K) {
                tdm_load_tile((unsigned)(uintptr_t)&sA[ibuf ^ 1][0],
                              A + mblock * K + (k0 + 32), 256, K);
                tdm_load_tile((unsigned)(uintptr_t)&sB[ibuf ^ 1][0],
                              WT + nblock * K + (k0 + 32), 128, K);
                tdm_wait<2>();   // oldest 2 (current buffer) complete
            } else {
                tdm_wait<0>();
            }
        }
        __syncthreads();
        const __bf16* tileA = &sA[ibuf][0];
        const __bf16* tileB = &sB[ibuf][0];
#else
        // no-TDM fallback: cooperative direct copy into LDS buffer 0
        {
            const __bf16* gA = A + mblock * K + k0;
            const __bf16* gB = WT + nblock * K + k0;
            for (int r = wave; r < 256; r += 8) {
                sA[0][r * LDSPITCH_ + lane] = gA[(long)r * K + lane];
            }
            for (int r = wave; r < 128; r += 8) {
                sB[0][r * LDSPITCH_ + lane] = gB[(long)r * K + lane];
            }
        }
        __syncthreads();
        const __bf16* tileA = &sA[0][0];
        const __bf16* tileB = &sB[0][0];
#endif
        // fragment loads from LDS
        v16bf af[4];
        #pragma unroll
        for (int i = 0; i < 4; ++i) {
            const int row = wm * 64 + i * 16 + l15;
            const __bf16* ar = tileA + row * LDSPITCH_ + hh * 8;
            v8bf lo = *(const v8bf*)(ar);
            v8bf hi = *(const v8bf*)(ar + 16);
            #pragma unroll
            for (int e = 0; e < 8; ++e) { af[i][e] = lo[e]; af[i][8 + e] = hi[e]; }
        }
        v16bf bfrag[4];
        #pragma unroll
        for (int j = 0; j < 4; ++j) {
            const int row = wn * 64 + j * 16 + l15;
            const __bf16* br = tileB + row * LDSPITCH_ + hh * 8;
            v8bf lo = *(const v8bf*)(br);
            v8bf hi = *(const v8bf*)(br + 16);
            #pragma unroll
            for (int e = 0; e < 8; ++e) { bfrag[j][e] = lo[e]; bfrag[j][8 + e] = hi[e]; }
        }
        #pragma unroll
        for (int i = 0; i < 4; ++i)
            #pragma unroll
            for (int j = 0; j < 4; ++j)
                acc[i][j] = __builtin_amdgcn_wmma_f32_16x16x32_bf16(
                    false, af[i], false, bfrag[j], (short)0, acc[i][j], false, false);
        __syncthreads();
        ibuf ^= 1;
    }

    // epilogue: bias (+gelu), store.  D layout: lane l15 = n, c[r] row = r+8*hh
    const long mbase = mblock + wm * 64;
    const long nbase = nblock + wn * 64;
    #pragma unroll
    for (int j = 0; j < 4; ++j) {
        const long n = nbase + j * 16 + l15;
        const float bval = bias[n];
        #pragma unroll
        for (int i = 0; i < 4; ++i) {
            #pragma unroll
            for (int r = 0; r < 8; ++r) {
                const long m = mbase + i * 16 + r + 8 * hh;
                float v = acc[i][j][r] + bval;
                if constexpr (MODE == 2)
                    v = 0.5f * v * (1.0f + erff(v * 0.70710678118654752f));
                if constexpr (MODE == 0)
                    ((float*)Cout)[m * N + n] = v;
                else
                    ((__bf16*)Cout)[m * N + n] = (__bf16)v;
            }
        }
    }
}

// ---------------------------------------------------------------------------
// Causal attention, flash-style.  One wave per (b, head, 16-query tile).
// S^T = K_tile @ Q^T  -> lane = query; softmax stats via one shfl_xor(16);
// exp'd scores repack register-locally into the P A-fragment; O += P @ V.
// ---------------------------------------------------------------------------
__global__ void attn_kernel(const __bf16* __restrict__ Q,
                            const __bf16* __restrict__ Km,
                            const __bf16* __restrict__ VT,
                            __bf16* __restrict__ O) {
    const int bh    = blockIdx.x;
    const int b     = bh / NH_;
    const int head  = bh - b * NH_;
    const int qbase = blockIdx.y * 16;
    const int lane  = threadIdx.x & 31;
    const int hh    = lane >> 4;
    const int l15   = lane & 15;
    const float scale = 0.125f;
    const float NEGINF = -__builtin_inff();

    const __bf16* qrow = Q + ((long)(b * S_ + qbase + l15)) * H_ + head * D_;
    v16bf qb[2];
    qb[0] = *(const v16bf*)(qrow + hh * 16);
    qb[1] = *(const v16bf*)(qrow + 32 + hh * 16);

    v8f o[4];
    #pragma unroll
    for (int s4 = 0; s4 < 4; ++s4)
        o[s4] = (v8f){0.f, 0.f, 0.f, 0.f, 0.f, 0.f, 0.f, 0.f};
    float mrun = NEGINF, lrun = 0.f;

    const int kend = qbase + 16;
    for (int kb0 = 0; kb0 < kend; kb0 += 32) {
        float p[2][8];
        float chmax = NEGINF;
        #pragma unroll
        for (int kc = 0; kc < 2; ++kc) {
            const int kt = kb0 + kc * 16;
            if (kt < kend) {               // uniform branch: EXEC stays full
                const __bf16* krow = Km + ((long)(b * S_ + kt + l15)) * H_ + head * D_;
                v16bf ka[2];
                #pragma unroll
                for (int f = 0; f < 2; ++f) {
                    v8bf lo = *(const v8bf*)(krow + f * 32 + hh * 16);
                    v8bf hi = *(const v8bf*)(krow + f * 32 + hh * 16 + 8);
                    #pragma unroll
                    for (int e = 0; e < 8; ++e) { ka[f][e] = lo[e]; ka[f][8 + e] = hi[e]; }
                }
                v8f sc = (v8f){0.f, 0.f, 0.f, 0.f, 0.f, 0.f, 0.f, 0.f};
                sc = __builtin_amdgcn_wmma_f32_16x16x32_bf16(false, ka[0], false, qb[0],
                                                             (short)0, sc, false, false);
                sc = __builtin_amdgcn_wmma_f32_16x16x32_bf16(false, ka[1], false, qb[1],
                                                             (short)0, sc, false, false);
                #pragma unroll
                for (int r = 0; r < 8; ++r) {
                    const int kg = kt + r + 8 * hh;
                    const int qg = qbase + l15;
                    const float v = (kg <= qg) ? sc[r] * scale : NEGINF;
                    p[kc][r] = v;
                    chmax = fmaxf(chmax, v);
                }
            } else {
                #pragma unroll
                for (int r = 0; r < 8; ++r) p[kc][r] = NEGINF;
            }
        }
        chmax = fmaxf(chmax, __shfl_xor(chmax, 16, 32));
        const float mnew = fmaxf(mrun, chmax);
        const float corr = expf(mrun - mnew);
        float lsum = 0.f;
        #pragma unroll
        for (int kc = 0; kc < 2; ++kc)
            #pragma unroll
            for (int r = 0; r < 8; ++r) {
                p[kc][r] = expf(p[kc][r] - mnew);
                lsum += p[kc][r];
            }
        lsum += __shfl_xor(lsum, 16, 32);
        lrun = lrun * corr + lsum;
        mrun = mnew;
        #pragma unroll
        for (int r = 0; r < 8; ++r) {
            const float cr = __shfl(corr, r + 8 * hh, 32);
            #pragma unroll
            for (int s4 = 0; s4 < 4; ++s4) o[s4][r] *= cr;
        }
        v16bf pf;
        #pragma unroll
        for (int i = 0; i < 8; ++i) {
            const int kc = (i < 4) ? 0 : 1;
            const int r2 = 2 * (i & 3);
            pf[2 * i]     = (__bf16)p[kc][r2];
            pf[2 * i + 1] = (__bf16)p[kc][r2 + 1];
        }
        #pragma unroll
        for (int s4 = 0; s4 < 4; ++s4) {
            int koff = kb0 + hh * 8;
            int koff2 = kb0 + 16 + hh * 8;
            if (koff2 >= kend) koff2 = koff;  // masked half: p==0, avoid OOB read
            const __bf16* vrow =
                VT + (((long)(b * NH_ + head)) * D_ + s4 * 16 + l15) * S_;
            v8bf vlo = *(const v8bf*)(vrow + koff);
            v8bf vhi = *(const v8bf*)(vrow + koff2);
            v16bf vb;
            #pragma unroll
            for (int e = 0; e < 8; ++e) { vb[e] = vlo[e]; vb[8 + e] = vhi[e]; }
            o[s4] = __builtin_amdgcn_wmma_f32_16x16x32_bf16(false, pf, false, vb,
                                                            (short)0, o[s4], false, false);
        }
    }
    #pragma unroll
    for (int r = 0; r < 8; ++r) {
        const float lr = __shfl(lrun, r + 8 * hh, 32);
        const float inv = 1.0f / lr;
        const long row = (long)b * S_ + qbase + r + 8 * hh;
        #pragma unroll
        for (int s4 = 0; s4 < 4; ++s4)
            O[row * H_ + head * D_ + s4 * 16 + l15] = (__bf16)(o[s4][r] * inv);
    }
}

// ---------------------------------------------------------------------------
// hF = LN(hF + dF); refresh bf16 mirror.
// ---------------------------------------------------------------------------
__global__ void residual_ln_kernel(float* __restrict__ hF,
                                   const float* __restrict__ dF,
                                   __bf16* __restrict__ hB) {
    const long row = blockIdx.x;
    const int tid = threadIdx.x;
    float x[4];
    #pragma unroll
    for (int u = 0; u < 4; ++u) {
        const int j = tid + u * 256;
        x[u] = hF[row * H_ + j] + dF[row * H_ + j];
    }
    float s = x[0] + x[1] + x[2] + x[3];
    const float mean = block_sum256(s) * (1.0f / H_);
    float vs = 0.f;
    #pragma unroll
    for (int u = 0; u < 4; ++u) { float d = x[u] - mean; vs += d * d; }
    const float var = block_sum256(vs) * (1.0f / H_);
    const float rs  = rsqrtf(var + LN_EPS_);
    #pragma unroll
    for (int u = 0; u < 4; ++u) {
        const int j = tid + u * 256;
        const float o = (x[u] - mean) * rs;
        hF[row * H_ + j] = o;
        hB[row * H_ + j] = (__bf16)o;
    }
}

// ---------------------------------------------------------------------------
// Prediction heads.
// ---------------------------------------------------------------------------
__global__ void heads_kernel(const float* __restrict__ hF,
                             const float* __restrict__ Wps, const float* __restrict__ bps,
                             const float* __restrict__ Wpa, const float* __restrict__ bpa,
                             const float* __restrict__ Wpr, const float* __restrict__ bpr,
                             float* __restrict__ outS, float* __restrict__ outA,
                             float* __restrict__ outR) {
    const int bt = blockIdx.x;
    const int b = bt / T_, t = bt - b * T_;
    const int tid = threadIdx.x;
    const float* hs = hF + ((long)b * S_ + 3 * t + 1) * H_;
    const float* ha = hF + ((long)b * S_ + 3 * t + 2) * H_;
    for (int o = 0; o < 17; ++o) {
        float p = 0.f;
        for (int j = tid; j < H_; j += 256) p += ha[j] * Wps[j * 17 + o];
        p = block_sum256(p);
        if (tid == 0) outS[(long)bt * 17 + o] = p + bps[o];
    }
    for (int o = 0; o < 18; ++o) {
        float p = 0.f;
        for (int j = tid; j < H_; j += 256) p += hs[j] * Wpa[j * 18 + o];
        p = block_sum256(p);
        if (tid == 0) outA[(long)bt * 18 + o] = p + bpa[o];
    }
    {
        float p = 0.f;
        for (int j = tid; j < H_; j += 256) p += ha[j] * Wpr[j];
        p = block_sum256(p);
        if (tid == 0) outR[bt] = p + bpr[0];
    }
}

// ---------------------------------------------------------------------------
extern "C" void kernel_launch(void* const* d_in, const int* in_sizes, int n_in,
                              void* d_out, int out_size, void* d_ws, size_t ws_size,
                              hipStream_t stream) {
    (void)in_sizes; (void)n_in; (void)out_size; (void)ws_size;
    const float* states = (const float*)d_in[0];
    const float* rtg    = (const float*)d_in[1];
    const int*   tsteps = (const int*)d_in[2];
    const int*   acts   = (const int*)d_in[3];
    const float* tt   = (const float*)d_in[5];
    const float* at   = (const float*)d_in[6];
    const float* Wrtg = (const float*)d_in[7];
    const float* brtg = (const float*)d_in[8];
    const float* Wst  = (const float*)d_in[9];
    const float* bst  = (const float*)d_in[10];
    const float* Wq = (const float*)d_in[11]; const float* bq = (const float*)d_in[12];
    const float* Wk = (const float*)d_in[13]; const float* bk = (const float*)d_in[14];
    const float* Wv = (const float*)d_in[15]; const float* bv = (const float*)d_in[16];
    const float* Wo = (const float*)d_in[17]; const float* bo = (const float*)d_in[18];
    const float* W1 = (const float*)d_in[19]; const float* b1 = (const float*)d_in[20];
    const float* W2 = (const float*)d_in[21]; const float* b2 = (const float*)d_in[22];
    const float* Wpr = (const float*)d_in[23]; const float* bpr = (const float*)d_in[24];
    const float* Wps = (const float*)d_in[25]; const float* bps = (const float*)d_in[26];
    const float* Wpa = (const float*)d_in[27]; const float* bpa = (const float*)d_in[28];

    char* ws = (char*)d_ws;
    auto alloc = [&](size_t bytes) {
        char* p = ws;
        ws += (bytes + 255) & ~(size_t)255;
        return p;
    };
    float*  hF   = (float*) alloc((size_t)MROWS_ * H_ * 4);
    __bf16* hB   = (__bf16*)alloc((size_t)MROWS_ * H_ * 2);
    __bf16* qB   = (__bf16*)alloc((size_t)MROWS_ * H_ * 2);
    __bf16* kB   = (__bf16*)alloc((size_t)MROWS_ * H_ * 2);
    __bf16* vB   = (__bf16*)alloc((size_t)MROWS_ * H_ * 2);
    __bf16* vT   = (__bf16*)alloc((size_t)MROWS_ * H_ * 2);
    __bf16* oB   = (__bf16*)alloc((size_t)MROWS_ * H_ * 2);
    __bf16* mB   = (__bf16*)alloc((size_t)MROWS_ * FF_ * 2);
    float*  scrF = (float*) alloc((size_t)MROWS_ * H_ * 4);
    __bf16* wqT  = (__bf16*)alloc((size_t)H_ * H_ * 2);
    __bf16* wkT  = (__bf16*)alloc((size_t)H_ * H_ * 2);
    __bf16* wvT  = (__bf16*)alloc((size_t)H_ * H_ * 2);
    __bf16* woT  = (__bf16*)alloc((size_t)H_ * H_ * 2);
    __bf16* w1T  = (__bf16*)alloc((size_t)H_ * FF_ * 2);
    __bf16* w2T  = (__bf16*)alloc((size_t)FF_ * H_ * 2);

    float* outS = (float*)d_out;
    float* outA = outS + (long)B_ * T_ * 17;
    float* outR = outA + (long)B_ * T_ * 18;

    embed_ln_kernel<<<MROWS_, 256, 0, stream>>>(states, rtg, tsteps, acts, tt, at,
                                                Wrtg, brtg, Wst, bst, hF, hB);

    const dim3 gemmBlk(256);
    const dim3 gridN1024(H_ / 128, MROWS_ / 256);      // (8, 24)
    const dim3 gridN4096(FF_ / 128, MROWS_ / 256);     // (32, 24)
    const int ctHH = (H_ * H_ + 255) / 256;
    const int ctHF = (H_ * FF_ + 255) / 256;

    for (int l = 0; l < L_; ++l) {
        const long wHH = (long)l * H_ * H_;
        const long wHF = (long)l * H_ * FF_;
        conv_transpose_kernel<<<ctHH, 256, 0, stream>>>(Wq + wHH, wqT, H_, H_);
        conv_transpose_kernel<<<ctHH, 256, 0, stream>>>(Wk + wHH, wkT, H_, H_);
        conv_transpose_kernel<<<ctHH, 256, 0, stream>>>(Wv + wHH, wvT, H_, H_);
        conv_transpose_kernel<<<ctHH, 256, 0, stream>>>(Wo + wHH, woT, H_, H_);
        conv_transpose_kernel<<<ctHF, 256, 0, stream>>>(W1 + wHF, w1T, H_, FF_);
        conv_transpose_kernel<<<ctHF, 256, 0, stream>>>(W2 + wHF, w2T, FF_, H_);

        gemm_wmma_kernel<1><<<gridN1024, gemmBlk, 0, stream>>>(hB, wqT, bq + (long)l * H_, qB, MROWS_, H_, H_);
        gemm_wmma_kernel<1><<<gridN1024, gemmBlk, 0, stream>>>(hB, wkT, bk + (long)l * H_, kB, MROWS_, H_, H_);
        gemm_wmma_kernel<1><<<gridN1024, gemmBlk, 0, stream>>>(hB, wvT, bv + (long)l * H_, vB, MROWS_, H_, H_);
        transpose_v_kernel<<<(MROWS_ * H_ + 255) / 256, 256, 0, stream>>>(vB, vT);

        attn_kernel<<<dim3(B_ * NH_, S_ / 16), 32, 0, stream>>>(qB, kB, vT, oB);

        gemm_wmma_kernel<0><<<gridN1024, gemmBlk, 0, stream>>>(oB, woT, bo + (long)l * H_, scrF, MROWS_, H_, H_);
        residual_ln_kernel<<<MROWS_, 256, 0, stream>>>(hF, scrF, hB);

        gemm_wmma_kernel<2><<<gridN4096, gemmBlk, 0, stream>>>(hB, w1T, b1 + (long)l * FF_, mB, MROWS_, FF_, H_);
        gemm_wmma_kernel<0><<<gridN1024, gemmBlk, 0, stream>>>(mB, w2T, b2 + (long)l * H_, scrF, MROWS_, H_, FF_);
        residual_ln_kernel<<<MROWS_, 256, 0, stream>>>(hF, scrF, hB);
    }

    heads_kernel<<<B_ * T_, 256, 0, stream>>>(hF, Wps, bps, Wpa, bpa, Wpr, bpr,
                                              outS, outA, outR);
}
